// IntGNN_13194139533877
// MI455X (gfx1250) — compile-verified, matrix operand order checked by reading, not run on hardware
//
#include <hip/hip_runtime.h>
#include <hip/hip_bf16.h>
#include <math.h>

typedef __attribute__((ext_vector_type(16))) _Float16 v16h;
typedef __attribute__((ext_vector_type(8)))  float    v8f;

#define HID   64
#define HEADS 4
#define CDIM  64
#define HC    256   // HEADS*CDIM
#define EA_D  11
#define NEG_SLOPE 0.2f

// ---------------- helpers ----------------

__device__ __forceinline__ void atomicMaxF(float* addr, float v) {
  // works for mixed-sign floats given -inf init
  if (v >= 0.0f) atomicMax((int*)addr, __float_as_int(v));
  else           atomicMin((unsigned int*)addr, __float_as_uint(v));
}

__global__ void k_fill(float* p, float v, int n) {
  int i = blockIdx.x * blockDim.x + threadIdx.x;
  if (i < n) p[i] = v;
}

// ---------------- stage 0: self-loop attr = mean of incoming edge_attr ----------------

__global__ void k_loop_accum(const int* ei, int E, const float* edge_attr,
                             float* cnt, float* loop_sum) {
  int e = blockIdx.x * blockDim.x + threadIdx.x;
  if (e >= E) return;
  int d = ei[E + e];  // dst row
  atomicAdd(&cnt[d], 1.0f);
  #pragma unroll
  for (int k = 0; k < EA_D; ++k)
    atomicAdd(&loop_sum[d * EA_D + k], edge_attr[e * EA_D + k]);
}

__global__ void k_loop_div(float* loop_attr, const float* cnt, int N) {
  int n = blockIdx.x * blockDim.x + threadIdx.x;
  if (n >= N) return;
  float inv = 1.0f / fmaxf(cnt[n], 1.0f);
  #pragma unroll
  for (int k = 0; k < EA_D; ++k) loop_attr[n * EA_D + k] *= inv;
}

// ---------------- stage 1: node encoder -> h (f16) ----------------

__global__ void k_encoder(const float* x, const int* node_type, const float* type_emb,
                          const float* enc_W, const float* enc_b,
                          _Float16* h16, int N) {
  int n = blockIdx.x * blockDim.x + threadIdx.x;
  if (n >= N) return;
  float in[10];
  #pragma unroll
  for (int k = 0; k < 6; ++k) in[k] = x[n * 6 + k];
  int t = node_type[n];
  #pragma unroll
  for (int k = 0; k < 4; ++k) in[6 + k] = type_emb[t * 4 + k];
  for (int j = 0; j < HID; ++j) {
    float s = enc_b[j];
    #pragma unroll
    for (int k = 0; k < 10; ++k) s = fmaf(in[k], enc_W[k * HID + j], s);
    h16[n * HID + j] = (_Float16)fmaxf(s, 0.0f);
  }
}

// Convert f32 weights [HID, HC] -> f16 transposed-K layout Wt[nn*HID + k]
// so that WMMA B-fragment K-pairs (k, k+1) at a fixed column nn are adjacent.
__global__ void k_cvt_w(const float* Wl, const float* Wr, _Float16* wlT, _Float16* wrT) {
  int i = blockIdx.x * blockDim.x + threadIdx.x;
  if (i >= HID * HC) return;
  int k = i / HC, nn = i % HC;
  wlT[nn * HID + k] = (_Float16)Wl[i];
  wrT[nn * HID + k] = (_Float16)Wr[i];
}

// ---------------- stage 2: WMMA GEMM  out[N,256] = h16[N,64] @ W[64,256] + bias ----------------
// one wave32 per 16x16 output tile; K=64 in two 16x16x32 f16 WMMA steps.
// A row-major [N,64]; B given transposed as WT[col*64 + k] so fragments load as b128.

__global__ void k_gemm_wmma(const _Float16* __restrict__ A, const _Float16* __restrict__ WT,
                            const float* __restrict__ bias, float* __restrict__ out, int nrows) {
  int wave = (blockIdx.x * blockDim.x + threadIdx.x) >> 5;
  int lane = threadIdx.x & 31;
  const int ntn = HC / 16;             // 16 tiles along N
  int mt = wave / ntn, nt = wave % ntn;
  if (mt * 16 >= nrows) return;
  int half = lane >> 4;                // 0 or 1
  int m    = mt * 16 + (lane & 15);    // A row
  if (m >= nrows) m = nrows - 1;
  int nn   = nt * 16 + (lane & 15);    // B/D column
  v8f c = {};
  #pragma unroll
  for (int kk = 0; kk < HID; kk += 32) {
    v16h a, b;
    #pragma unroll
    for (int v = 0; v < 8; ++v) {
      int kb = kk + ((v >> 2) << 4) + (half << 3) + ((v & 3) << 1);
      a[2 * v]     = A[m * HID + kb];
      a[2 * v + 1] = A[m * HID + kb + 1];
      b[2 * v]     = WT[nn * HID + kb];
      b[2 * v + 1] = WT[nn * HID + kb + 1];
    }
    c = __builtin_amdgcn_wmma_f32_16x16x32_f16(false, a, false, b, (short)0, c, false, false);
  }
  #pragma unroll
  for (int r = 0; r < 8; ++r) {
    int mm = mt * 16 + r + half * 8;   // D: VGPR r -> M=r (lanes 0-15), M=r+8 (lanes 16-31)
    out[mm * HC + nn] = c[r] + bias[nn];
  }
}

// ---------------- stage 3a: per-edge attention logits + segment max ----------------
// one wave per (virtual) edge; lane covers 8 channels; lanes [8h,8h+8) = head h

__global__ void k_logits(const int* ei, int E, int Etot,
                         const float* edge_attr, const float* loop_attr,
                         const float* __restrict__ xl, const float* __restrict__ xr,
                         const float* __restrict__ We, const float* __restrict__ att,
                         float* logits, float* lmax) {
  int e    = (blockIdx.x * blockDim.x + threadIdx.x) >> 5;
  int lane = threadIdx.x & 31;
  if (e >= Etot) return;
  int s, d; const float* eap;
  if (e < E) { s = ei[e]; d = ei[E + e]; eap = edge_attr + (size_t)e * EA_D; }
  else       { s = e - E; d = e - E;     eap = loop_attr + (size_t)(e - E) * EA_D; }
  float eav[EA_D];
  #pragma unroll
  for (int k = 0; k < EA_D; ++k) eav[k] = eap[k];
  float lsum = 0.0f;
  #pragma unroll
  for (int i = 0; i < 8; ++i) {
    int ch = lane * 8 + i;             // head = ch>>6, att flat index == ch
    float ee = 0.0f;
    #pragma unroll
    for (int k = 0; k < EA_D; ++k) ee = fmaf(eav[k], We[k * HC + ch], ee);
    float mv = xl[(size_t)s * HC + ch] + xr[(size_t)d * HC + ch] + ee;
    mv = (mv > 0.0f) ? mv : NEG_SLOPE * mv;
    lsum = fmaf(mv, att[ch], lsum);
  }
  // reduce over the 8 lanes of each head group
  lsum += __shfl_xor(lsum, 1, 8);
  lsum += __shfl_xor(lsum, 2, 8);
  lsum += __shfl_xor(lsum, 4, 8);
  if ((lane & 7) == 0) {
    int h = lane >> 3;
    logits[(size_t)e * HEADS + h] = lsum;
    atomicMaxF(&lmax[(size_t)d * HEADS + h], lsum);
  }
}

// ---------------- stage 3b: exp + segment sum ----------------

__global__ void k_expsum(const int* ei, int E, int Etot, float* logits,
                         const float* lmax, float* denom) {
  int i = blockIdx.x * blockDim.x + threadIdx.x;
  if (i >= Etot * HEADS) return;
  int e = i >> 2, h = i & 3;
  int d = (e < E) ? ei[E + e] : (e - E);
  float a = __expf(logits[i] - lmax[d * HEADS + h]);
  logits[i] = a;
  atomicAdd(&denom[d * HEADS + h], a);
}

// ---------------- stage 3c: normalize alpha (fold head-mean 1/4) ----------------

__global__ void k_norm(const int* ei, int E, int Etot, float* alpha, const float* denom) {
  int i = blockIdx.x * blockDim.x + threadIdx.x;
  if (i >= Etot * HEADS) return;
  int e = i >> 2, h = i & 3;
  int d = (e < E) ? ei[E + e] : (e - E);
  alpha[i] = alpha[i] / (denom[d * HEADS + h] + 1e-16f) * 0.25f;
}

// ---------------- stage 3d: weighted aggregation, head-mean folded in ----------------
// one wave per edge; lane handles channels c=lane and c=lane+32

__global__ void k_aggregate(const int* ei, int E, int Etot, const float* __restrict__ alpha,
                            const float* __restrict__ xl, float* hacc) {
  int e    = (blockIdx.x * blockDim.x + threadIdx.x) >> 5;
  int lane = threadIdx.x & 31;
  if (e >= Etot) return;
  int s, d;
  if (e < E) { s = ei[e]; d = ei[E + e]; } else { s = e - E; d = e - E; }
  float a0 = alpha[(size_t)e * HEADS + 0];
  float a1 = alpha[(size_t)e * HEADS + 1];
  float a2 = alpha[(size_t)e * HEADS + 2];
  float a3 = alpha[(size_t)e * HEADS + 3];
  const float* xs = xl + (size_t)s * HC;
  #pragma unroll
  for (int half = 0; half < 2; ++half) {
    int c = lane + half * 32;
    float acc = a0 * xs[0 * CDIM + c] + a1 * xs[1 * CDIM + c]
              + a2 * xs[2 * CDIM + c] + a3 * xs[3 * CDIM + c];
    atomicAdd(&hacc[(size_t)d * CDIM + c], acc);
  }
}

// ---------------- stage 4: hconv = relu(hacc + bias); global mean pool accumulate ----------------

__global__ void k_hconv_pool(const float* hacc, const float* conv_bias, const int* batch,
                             float* pooled, float* gcnt, int N) {
  int n = blockIdx.x * blockDim.x + threadIdx.x;
  if (n >= N) return;
  int b = batch[n];
  #pragma unroll 8
  for (int c = 0; c < CDIM; ++c) {
    float v = fmaxf(hacc[(size_t)n * CDIM + c] + conv_bias[c], 0.0f);
    atomicAdd(&pooled[b * CDIM + c], v);
  }
  atomicAdd(&gcnt[b], 1.0f);
}

// ---------------- stage 5: per-graph head MLP ----------------

__global__ void k_head(const float* pooled, const float* gcnt, const float* gf,
                       const float* gW, const float* gb,
                       const float* h1W, const float* h1b,
                       const float* h2W, const float* h2b,
                       float* out, int G) {
  __shared__ float comb[64 * 128];
  int g = threadIdx.x;
  if (g >= G) return;
  float* cb = comb + g * 128;
  float inv = 1.0f / fmaxf(gcnt[g], 1.0f);
  for (int c = 0; c < CDIM; ++c) cb[c] = pooled[g * CDIM + c] * inv;
  for (int c = 0; c < HID; ++c) {
    float s = gb[c];
    #pragma unroll
    for (int k = 0; k < 6; ++k) s = fmaf(gf[g * 6 + k], gW[k * HID + c], s);
    cb[64 + c] = fmaxf(s, 0.0f);
  }
  float o = h2b[0];
  for (int j = 0; j < HID; ++j) {
    float s = h1b[j];
    for (int k = 0; k < 128; ++k) s = fmaf(cb[k], h1W[k * HID + j], s);
    o = fmaf(fmaxf(s, 0.0f), h2W[j], o);
  }
  out[g] = o;
}

// ---------------- launch ----------------

extern "C" void kernel_launch(void* const* d_in, const int* in_sizes, int n_in,
                              void* d_out, int out_size, void* d_ws, size_t ws_size,
                              hipStream_t stream) {
  const float* x         = (const float*)d_in[0];
  const int*   node_type = (const int*)  d_in[1];
  const int*   ei        = (const int*)  d_in[2];
  const float* edge_attr = (const float*)d_in[3];
  const int*   batch     = (const int*)  d_in[4];
  const float* gf        = (const float*)d_in[5];
  const float* type_emb  = (const float*)d_in[6];
  const float* enc_W     = (const float*)d_in[7];
  const float* enc_b     = (const float*)d_in[8];
  const float* Wl        = (const float*)d_in[9];
  const float* bl        = (const float*)d_in[10];
  const float* Wr        = (const float*)d_in[11];
  const float* br        = (const float*)d_in[12];
  const float* We        = (const float*)d_in[13];
  const float* att       = (const float*)d_in[14];
  const float* conv_bias = (const float*)d_in[15];
  const float* gW        = (const float*)d_in[16];
  const float* gb        = (const float*)d_in[17];
  const float* h1W       = (const float*)d_in[18];
  const float* h1b       = (const float*)d_in[19];
  const float* h2W       = (const float*)d_in[20];
  const float* h2b       = (const float*)d_in[21];
  float* out = (float*)d_out;

  const int N = in_sizes[0] / 6;
  const int E = in_sizes[3] / EA_D;
  const int G = in_sizes[5] / 6;
  const int Etot = E + N;

  // workspace layout
  char* p = (char*)d_ws;
  _Float16* h16  = (_Float16*)p; p += (size_t)N * HID * sizeof(_Float16);
  _Float16* wlT  = (_Float16*)p; p += (size_t)HID * HC * sizeof(_Float16);
  _Float16* wrT  = (_Float16*)p; p += (size_t)HID * HC * sizeof(_Float16);
  p = (char*)(((uintptr_t)p + 255) & ~(uintptr_t)255);
  float* xl        = (float*)p; p += (size_t)N * HC * sizeof(float);
  float* xr        = (float*)p; p += (size_t)N * HC * sizeof(float);
  float* cnt       = (float*)p; p += (size_t)N * sizeof(float);
  float* loop_attr = (float*)p; p += (size_t)N * EA_D * sizeof(float);
  float* logits    = (float*)p; p += (size_t)Etot * HEADS * sizeof(float);
  float* lmax      = (float*)p; p += (size_t)N * HEADS * sizeof(float);
  float* denom     = (float*)p; p += (size_t)N * HEADS * sizeof(float);
  float* hacc      = (float*)p; p += (size_t)N * CDIM * sizeof(float);
  float* pooled    = (float*)p; p += (size_t)G * CDIM * sizeof(float);
  float* gcnt      = (float*)p; p += (size_t)G * sizeof(float);
  (void)ws_size; (void)n_in; (void)out_size;

  const int B = 256;
  #define GRD(n) (((n) + B - 1) / B)

  // init accumulators (ws is poisoned)
  k_fill<<<GRD(N),        B, 0, stream>>>(cnt, 0.0f, N);
  k_fill<<<GRD(N * EA_D), B, 0, stream>>>(loop_attr, 0.0f, N * EA_D);
  k_fill<<<GRD(N * HEADS),B, 0, stream>>>(lmax, -INFINITY, N * HEADS);
  k_fill<<<GRD(N * HEADS),B, 0, stream>>>(denom, 0.0f, N * HEADS);
  k_fill<<<GRD(N * CDIM), B, 0, stream>>>(hacc, 0.0f, N * CDIM);
  k_fill<<<GRD(G * CDIM), B, 0, stream>>>(pooled, 0.0f, G * CDIM);
  k_fill<<<GRD(G),        B, 0, stream>>>(gcnt, 0.0f, G);

  // self-loop mean edge attr
  k_loop_accum<<<GRD(E), B, 0, stream>>>(ei, E, edge_attr, cnt, loop_attr);
  k_loop_div  <<<GRD(N), B, 0, stream>>>(loop_attr, cnt, N);

  // encoder + weight conversion (f16, K-contiguous transpose)
  k_encoder<<<GRD(N), B, 0, stream>>>(x, node_type, type_emb, enc_W, enc_b, h16, N);
  k_cvt_w  <<<GRD(HID * HC), B, 0, stream>>>(Wl, Wr, wlT, wrT);

  // WMMA GEMMs: xl = h@Wl+bl, xr = h@Wr+br
  int mtiles = (N + 15) / 16;
  int waves  = mtiles * (HC / 16);
  int gemm_blocks = (waves * 32 + B - 1) / B;
  k_gemm_wmma<<<gemm_blocks, B, 0, stream>>>(h16, wlT, bl, xl, N);
  k_gemm_wmma<<<gemm_blocks, B, 0, stream>>>(h16, wrT, br, xr, N);

  // attention: logits + segment softmax + aggregation
  int edge_blocks = ((size_t)Etot * 32 + B - 1) / B;
  k_logits   <<<edge_blocks, B, 0, stream>>>(ei, E, Etot, edge_attr, loop_attr,
                                             xl, xr, We, att, logits, lmax);
  k_expsum   <<<GRD(Etot * HEADS), B, 0, stream>>>(ei, E, Etot, logits, lmax, denom);
  k_norm     <<<GRD(Etot * HEADS), B, 0, stream>>>(ei, E, Etot, logits, denom);
  k_aggregate<<<edge_blocks, B, 0, stream>>>(ei, E, Etot, logits, xl, hacc);

  // conv output + pooling + head
  k_hconv_pool<<<GRD(N), B, 0, stream>>>(hacc, conv_bias, batch, pooled, gcnt, N);
  k_head<<<1, 64, 0, stream>>>(pooled, gcnt, gf, gW, gb, h1W, h1b, h2W, h2b, out, G);
  #undef GRD
}
